// bPC_Net_44959717655267
// MI455X (gfx1250) — matrix operand order repaired and checked by reading, not compiled, not run on hardware
//
#include <hip/hip_runtime.h>

typedef __bf16 bf16;
typedef __attribute__((ext_vector_type(16))) __bf16 v16bf;
typedef __attribute__((ext_vector_type(8)))  __bf16 v8bf;
typedef __attribute__((ext_vector_type(8)))  float  v8f;
typedef __attribute__((ext_vector_type(4)))  float  v4f;

#define BATCH   1024
#define NSTEPS  20
#define LR_X    0.05f
#define MOM     0.9f

static const int H_DIMS[5] = {784, 2048, 2048, 2048, 10};
static const int H_PD[5]   = {896, 2048, 2048, 2048, 128};

// ---------------------------------------------------------------------------
// small utility kernels
// ---------------------------------------------------------------------------
__global__ void k_zero(float* p, size_t n) {
    size_t i = (size_t)blockIdx.x * blockDim.x + threadIdx.x;
    if (i < n) p[i] = 0.0f;
}

// t = tanh(x) as bf16, zero-padded to Npad columns (used only for t0)
__global__ void k_tanh_bf(const float* __restrict__ x, int ldx,
                          bf16* __restrict__ t, int N, int Npad) {
    size_t i = (size_t)blockIdx.x * blockDim.x + threadIdx.x;
    size_t total = (size_t)BATCH * Npad;
    if (i >= total) return;
    int b = (int)(i / Npad), n = (int)(i % Npad);
    float v = 0.0f;
    if (n < N) v = tanhf(x[(size_t)b * ldx + n]);
    t[i] = (bf16)v;
}

__global__ void k_out(const float* __restrict__ x4, float* __restrict__ out) {
    int i = blockIdx.x * blockDim.x + threadIdx.x;
    if (i < BATCH * 10) {
        int b = i / 10, j = i % 10;
        out[i] = x4[(size_t)b * 128 + j];
    }
}

// batched fp32 -> zero-padded bf16 weight conversion (plain / transposed)
struct ConvDesc {
    const float* src; bf16* dst;
    int R, C, Rp, Cp, tr; size_t n;
};
struct ConvArgs { ConvDesc d[16]; };

__global__ void k_convert_batch(ConvArgs args) {
    const ConvDesc dd = args.d[blockIdx.z];
    size_t i = (size_t)blockIdx.x * blockDim.x + threadIdx.x;
    if (i >= dd.n) return;
    float v = 0.0f;
    if (dd.tr == 0) {
        int r = (int)(i / dd.Cp), c = (int)(i % dd.Cp);
        if (r < dd.R && c < dd.C) v = dd.src[(size_t)r * dd.C + c];
    } else {
        int c = (int)(i / dd.Rp), r = (int)(i % dd.Rp);
        if (r < dd.R && c < dd.C) v = dd.src[(size_t)r * dd.C + c];
    }
    dd.dst[i] = (bf16)v;
}

// ---------------------------------------------------------------------------
// WMMA bf16 NT-GEMM core: C[m,n] += sum_k A[m,k] * B[n,k]
//   256 threads (8 wave32), tile 128(M) x 128(N), K step 32,
//   double-buffered LDS: loads for tile i+1 staged in regs while WMMAs run
//   on tile i; one barrier per K-step. Safe because all K/32 counts are even.
// ---------------------------------------------------------------------------
template<bool AF32>
__device__ inline void gemm_accum(v8f acc[2][4],
        const void* __restrict__ Aptr, int ldA,
        const bf16* __restrict__ Bptr, int ldB, int K,
        int m0, int n0,
        bf16 (*As)[128][40], bf16 (*Bs)[128][40]) {
    const int tid  = threadIdx.x;
    const int lane = tid & 31;
    const int wid  = tid >> 5;
    const int wm   = (wid & 3) * 32;
    const int wn   = (wid >> 2) * 64;
    const int half = lane >> 4;
    const int r    = lane & 15;
    const int lrow = tid >> 1;          // 0..127 cooperative-load row
    const int lcol = (tid & 1) * 16;    // 0/16   cooperative-load col

    const float* gaf = (const float*)Aptr + (size_t)(m0 + lrow) * ldA + lcol;
    const bf16*  gab = (const bf16*) Aptr + (size_t)(m0 + lrow) * ldA + lcol;
    const bf16*  gb  = Bptr + (size_t)(n0 + lrow) * ldB + lcol;

    v8bf sa0, sa1, sb0, sb1;            // staged bf16 tiles
    v4f  sf0, sf1, sf2, sf3;            // staged fp32 A tile (AF32 path)

    auto load_tile = [&](int k0) {
        if (AF32) {
            const float* p = gaf + k0;
            sf0 = *(const v4f*)(p + 0);  sf1 = *(const v4f*)(p + 4);
            sf2 = *(const v4f*)(p + 8);  sf3 = *(const v4f*)(p + 12);
        } else {
            const bf16* p = gab + k0;
            sa0 = *(const v8bf*)(p);     sa1 = *(const v8bf*)(p + 8);
        }
        const bf16* q = gb + k0;
        sb0 = *(const v8bf*)(q);         sb1 = *(const v8bf*)(q + 8);
    };
    auto store_tile = [&](int buf) {
        if (AF32) {
            v8bf h0, h1;
            h0[0]=(bf16)sf0[0]; h0[1]=(bf16)sf0[1]; h0[2]=(bf16)sf0[2]; h0[3]=(bf16)sf0[3];
            h0[4]=(bf16)sf1[0]; h0[5]=(bf16)sf1[1]; h0[6]=(bf16)sf1[2]; h0[7]=(bf16)sf1[3];
            h1[0]=(bf16)sf2[0]; h1[1]=(bf16)sf2[1]; h1[2]=(bf16)sf2[2]; h1[3]=(bf16)sf2[3];
            h1[4]=(bf16)sf3[0]; h1[5]=(bf16)sf3[1]; h1[6]=(bf16)sf3[2]; h1[7]=(bf16)sf3[3];
            *(v8bf*)&As[buf][lrow][lcol]     = h0;
            *(v8bf*)&As[buf][lrow][lcol + 8] = h1;
        } else {
            *(v8bf*)&As[buf][lrow][lcol]     = sa0;
            *(v8bf*)&As[buf][lrow][lcol + 8] = sa1;
        }
        *(v8bf*)&Bs[buf][lrow][lcol]     = sb0;
        *(v8bf*)&Bs[buf][lrow][lcol + 8] = sb1;
    };

    const int niter = K >> 5;
    load_tile(0);
    store_tile(0);
    for (int i = 0; i < niter; ++i) {
        __syncthreads();
        const int cur = i & 1;
        const bool more = (i + 1) < niter;
        if (more) {
            load_tile((i + 1) << 5);                     // hide latency under WMMAs
            if (i + 2 < niter) {
                __builtin_prefetch(gb  + ((i + 2) << 5), 0, 1);
                if (AF32) __builtin_prefetch(gaf + ((i + 2) << 5), 0, 1);
                else      __builtin_prefetch(gab + ((i + 2) << 5), 0, 1);
            }
        }
        v16bf bfrag[4];
        #pragma unroll
        for (int ni = 0; ni < 4; ++ni) {
            int row = wn + ni * 16 + r;                  // B row = output column
            v8bf b0 = *(const v8bf*)&Bs[cur][row][half * 16];
            v8bf b1 = *(const v8bf*)&Bs[cur][row][half * 16 + 8];
            bfrag[ni] = __builtin_shufflevector(b0, b1,
                0,1,2,3,4,5,6,7,8,9,10,11,12,13,14,15);
        }
        #pragma unroll
        for (int mi = 0; mi < 2; ++mi) {
            int arow = wm + mi * 16 + r;
            v8bf a0 = *(const v8bf*)&As[cur][arow][half * 16];
            v8bf a1 = *(const v8bf*)&As[cur][arow][half * 16 + 8];
            v16bf afrag = __builtin_shufflevector(a0, a1,
                0,1,2,3,4,5,6,7,8,9,10,11,12,13,14,15);
            #pragma unroll
            for (int ni = 0; ni < 4; ++ni) {
                acc[mi][ni] = __builtin_amdgcn_wmma_f32_16x16x32_bf16(
                    false, afrag, false, bfrag[ni], (short)0, acc[mi][ni],
                    false, false);
            }
        }
        if (more) store_tile(1 - cur);
    }
}

// ---------------------------------------------------------------------------
// batched eps / forward-init GEMM:
//   out[b,n] = (xin ? xin[b,n] - acc : acc) for n<N, 0 in padding
//   optional fused tout = bf16(tanh(out)) over the full padded width
// ---------------------------------------------------------------------------
struct EpsDesc {
    const void* A; const bf16* B; const float* xin; float* out; bf16* tout;
    int ldA, ldB, K, ldx, Npad, N;
};
struct EpsArgs { EpsDesc d[8]; };

__global__ __launch_bounds__(256) void k_gemm_eps_batch(EpsArgs args) {
    const EpsDesc dd = args.d[blockIdx.z];
    const int n0 = blockIdx.x * 128;
    if (n0 >= dd.Npad) return;                 // uniform per-block early exit
    const int m0 = blockIdx.y * 128;

    __shared__ __align__(16) bf16 As[2][128][40];
    __shared__ __align__(16) bf16 Bs[2][128][40];
    v8f acc[2][4];
    #pragma unroll
    for (int mi = 0; mi < 2; ++mi)
        #pragma unroll
        for (int ni = 0; ni < 4; ++ni) acc[mi][ni] = (v8f)0.0f;

    gemm_accum<false>(acc, dd.A, dd.ldA, dd.B, dd.ldB, dd.K, m0, n0, As, Bs);

    const int tid = threadIdx.x, lane = tid & 31, wid = tid >> 5;
    const int wm = (wid & 3) * 32, wn = (wid >> 2) * 64;
    const int half = lane >> 4, r = lane & 15;
    #pragma unroll
    for (int mi = 0; mi < 2; ++mi)
        #pragma unroll
        for (int ni = 0; ni < 4; ++ni)
            #pragma unroll
            for (int j = 0; j < 8; ++j) {
                int mrow = m0 + wm + mi * 16 + half * 8 + j;
                int n    = n0 + wn + ni * 16 + r;
                float v = 0.0f;
                if (n < dd.N) {
                    float g = acc[mi][ni][j];
                    v = dd.xin ? (dd.xin[(size_t)mrow * dd.ldx + n] - g) : g;
                }
                size_t idx = (size_t)mrow * dd.Npad + n;
                dd.out[idx] = v;
                if (dd.tout) dd.tout[idx] = (bf16)tanhf(v);
            }
}

// ---------------------------------------------------------------------------
// batched propagation + fused state update:
//   acc = A1@B1^T (+ A2@B2^T)
//   delta = -(eg + ed) + (1 - tanh(x)^2) * acc ; m = MOM*m + delta
//   x += LR*m ; t = bf16(tanh(x_new))
// ---------------------------------------------------------------------------
struct PropDesc {
    const float* A1; const bf16* B1; const float* A2; const bf16* B2;
    const float* eg; const float* ed; float* x; float* m; bf16* tout;
    int ldA1, K1, ldA2, K2, Npad, N;
};
struct PropArgs { PropDesc d[4]; };

__global__ __launch_bounds__(256) void k_prop_batch(PropArgs args) {
    const PropDesc dd = args.d[blockIdx.z];
    const int n0 = blockIdx.x * 128;
    if (n0 >= dd.Npad) return;
    const int m0 = blockIdx.y * 128;

    __shared__ __align__(16) bf16 As[2][128][40];
    __shared__ __align__(16) bf16 Bs[2][128][40];
    v8f acc[2][4];
    #pragma unroll
    for (int mi = 0; mi < 2; ++mi)
        #pragma unroll
        for (int ni = 0; ni < 4; ++ni) acc[mi][ni] = (v8f)0.0f;

    gemm_accum<true>(acc, dd.A1, dd.ldA1, dd.B1, dd.K1, dd.K1, m0, n0, As, Bs);
    if (dd.A2)
        gemm_accum<true>(acc, dd.A2, dd.ldA2, dd.B2, dd.K2, dd.K2, m0, n0, As, Bs);

    const int tid = threadIdx.x, lane = tid & 31, wid = tid >> 5;
    const int wm = (wid & 3) * 32, wn = (wid >> 2) * 64;
    const int half = lane >> 4, r = lane & 15;
    #pragma unroll
    for (int mi = 0; mi < 2; ++mi)
        #pragma unroll
        for (int ni = 0; ni < 4; ++ni)
            #pragma unroll
            for (int j = 0; j < 8; ++j) {
                int mrow = m0 + wm + mi * 16 + half * 8 + j;
                int n    = n0 + wn + ni * 16 + r;
                if (n < dd.N) {
                    size_t idx = (size_t)mrow * dd.Npad + n;
                    float egv = dd.eg ? dd.eg[idx] : 0.0f;
                    float edv = dd.ed[idx];
                    float xv  = dd.x[idx];
                    float t   = tanhf(xv);
                    float delta = -(egv + edv) + (1.0f - t * t) * acc[mi][ni][j];
                    float mv = MOM * dd.m[idx] + delta;
                    dd.m[idx] = mv;
                    float xn = xv + LR_X * mv;
                    dd.x[idx] = xn;
                    dd.tout[idx] = (bf16)tanhf(xn);
                }
            }
}

// ---------------------------------------------------------------------------
// host orchestration
// ---------------------------------------------------------------------------
extern "C" void kernel_launch(void* const* d_in, const int* in_sizes, int n_in,
                              void* d_out, int out_size, void* d_ws, size_t ws_size,
                              hipStream_t stream) {
    const float* x_in = (const float*)d_in[0];
    const float* V[4] = {(const float*)d_in[1], (const float*)d_in[2],
                         (const float*)d_in[3], (const float*)d_in[4]};
    const float* W[4] = {(const float*)d_in[5], (const float*)d_in[6],
                         (const float*)d_in[7], (const float*)d_in[8]};
    (void)in_sizes; (void)n_in; (void)out_size;

    const int* D  = H_DIMS;
    const int* PD = H_PD;

    // ---- workspace layout ----
    char* base = (char*)d_ws;
    size_t off = 0;
    auto alloc = [&](size_t bytes) -> void* {
        void* p = base + off;
        off = (off + bytes + 255) & ~(size_t)255;
        return p;
    };
    bf16*  T[5];  for (int l = 0; l < 5; ++l) T[l]  = (bf16*)alloc((size_t)BATCH * PD[l] * sizeof(bf16));
    float* EG[4]; for (int l = 0; l < 4; ++l) EG[l] = (float*)alloc((size_t)BATCH * PD[l] * sizeof(float));
    float* ED[5]; for (int l = 1; l < 5; ++l) ED[l] = (float*)alloc((size_t)BATCH * PD[l] * sizeof(float));
    float* X[5];  for (int l = 1; l < 5; ++l) X[l]  = (float*)alloc((size_t)BATCH * PD[l] * sizeof(float));
    size_t mElems = (size_t)BATCH * (PD[1] + PD[2] + PD[3] + PD[4]);
    float* Mslab = (float*)alloc(mElems * sizeof(float));
    float* Mv[5];
    { size_t o = 0; for (int l = 1; l < 5; ++l) { Mv[l] = Mslab + o; o += (size_t)BATCH * PD[l]; } }
    bf16 *Vb[4], *VT[4], *Wb[4], *WT[4];
    for (int l = 0; l < 4; ++l) {
        size_t ne = (size_t)PD[l + 1] * PD[l];
        Vb[l] = (bf16*)alloc(ne * sizeof(bf16));
        VT[l] = (bf16*)alloc(ne * sizeof(bf16));
        Wb[l] = (bf16*)alloc(ne * sizeof(bf16));
        WT[l] = (bf16*)alloc(ne * sizeof(bf16));
    }
    if (off > ws_size) return;  // workspace too small: bail deterministically

    auto cdiv = [](size_t a, size_t b) { return (unsigned)((a + b - 1) / b); };

    // ---- init: zero momentum, batched weight conversion ----
    k_zero<<<cdiv(mElems, 256), 256, 0, stream>>>(Mslab, mElems);
    {
        ConvArgs ca{};
        unsigned maxb = 0;
        int z = 0;
        for (int l = 0; l < 4; ++l) {
            size_t ne = (size_t)PD[l + 1] * PD[l];
            ca.d[z++] = {V[l], Vb[l], D[l+1], D[l], PD[l+1], PD[l], 0, ne};
            ca.d[z++] = {V[l], VT[l], D[l+1], D[l], PD[l+1], PD[l], 1, ne};
            ca.d[z++] = {W[l], Wb[l], D[l], D[l+1], PD[l], PD[l+1], 0, ne};
            ca.d[z++] = {W[l], WT[l], D[l], D[l+1], PD[l], PD[l+1], 1, ne};
            if (cdiv(ne, 256) > maxb) maxb = cdiv(ne, 256);
        }
        dim3 g(maxb, 1, 16);
        k_convert_batch<<<g, 256, 0, stream>>>(ca);
    }

    // ---- forward init: x[l+1] = tanh(x[l]) @ V_l^T, fused t[l+1]=tanh(x[l+1]) ----
    k_tanh_bf<<<cdiv((size_t)BATCH * PD[0], 256), 256, 0, stream>>>(x_in, D[0], T[0], D[0], PD[0]);
    for (int l = 0; l < 4; ++l) {
        EpsArgs ea{};
        ea.d[0] = {T[l], Vb[l], nullptr, X[l + 1], T[l + 1],
                   PD[l], PD[l], PD[l], 0, PD[l + 1], D[l + 1]};
        dim3 g(PD[l + 1] / 128, BATCH / 128, 1);
        k_gemm_eps_batch<<<g, 256, 0, stream>>>(ea);
    }

    // ---- relaxation steps: 2 launches per step ----
    for (int s = 0; s < NSTEPS; ++s) {
        // all 8 eps GEMMs are independent -> one batched launch
        EpsArgs ea{};
        for (int l = 1; l < 5; ++l)   // eps_disc[l] = x[l] - tanh(x[l-1]) @ V^T
            ea.d[l - 1] = {T[l - 1], Vb[l - 1], X[l], ED[l], nullptr,
                           PD[l - 1], PD[l - 1], PD[l - 1], PD[l], PD[l], D[l]};
        for (int l = 0; l < 4; ++l) { // eps_gen[l] = x[l] - tanh(x[l+1]) @ W^T
            const float* xi = (l == 0) ? x_in : X[l];
            int ldx = (l == 0) ? D[0] : PD[l];
            ea.d[4 + l] = {T[l + 1], Wb[l], xi, EG[l], nullptr,
                           PD[l + 1], PD[l + 1], PD[l + 1], ldx, PD[l], D[l]};
        }
        dim3 ge(16, BATCH / 128, 8);
        k_gemm_eps_batch<<<ge, 256, 0, stream>>>(ea);

        // all 4 prop/update GEMMs are independent -> one batched launch
        PropArgs pa{};
        for (int l = 1; l < 5; ++l) {
            const float* A2  = (l < 4) ? ED[l + 1] : nullptr;
            const bf16*  B2  = (l < 4) ? VT[l]     : nullptr;
            int          K2  = (l < 4) ? PD[l + 1] : 32;   // unused when A2==null
            const float* egp = (l < 4) ? EG[l]     : nullptr;
            pa.d[l - 1] = {EG[l - 1], WT[l - 1], A2, B2, egp, ED[l],
                           X[l], Mv[l], T[l],
                           PD[l - 1], PD[l - 1], (l < 4) ? PD[l + 1] : 32, K2,
                           PD[l], D[l]};
        }
        dim3 gp(16, BATCH / 128, 4);
        k_prop_batch<<<gp, 256, 0, stream>>>(pa);
    }

    // ---- emit x[4][:, :10] ----
    k_out<<<cdiv((size_t)BATCH * 10, 256), 256, 0, stream>>>(X[4], (float*)d_out);
}
